// globalAttention_45586782880197
// MI455X (gfx1250) — matrix operand order repaired and checked by reading, not compiled, not run on hardware
//
#include <hip/hip_runtime.h>
#include <hip/hip_fp16.h>

typedef __attribute__((ext_vector_type(16))) _Float16 v16h;
typedef __attribute__((ext_vector_type(8)))  _Float16 v8h;
typedef __attribute__((ext_vector_type(8)))  float    v8f;
typedef __attribute__((ext_vector_type(4)))  unsigned v4u;
typedef __attribute__((ext_vector_type(8)))  unsigned v8u;

#define NTOK 5120      // t * (h/2)*(w/2) = 5*1024
#define DH   256       // p*p*c

// A-fragment K index for 16-bit 16x32 A matrix (contiguous order, matches B):
// lanes 0-15: K=0..15
// lanes16-31: K=16..31
__device__ __forceinline__ int a_frag_k(int grp, int e) {
    return grp * 16 + e;
}

// ---------------------------------------------------------------------------
// Kernel 1: depthwise 3x3 conv for q and k, fused unfold -> f16 token buffers
// grid: (bt=10, ch=64), block: 256 threads (16 pixels each)
// ---------------------------------------------------------------------------
__global__ __launch_bounds__(256) void qk_kernel(
    const float* __restrict__ x,
    const float* __restrict__ wq, const float* __restrict__ bq,
    const float* __restrict__ wk, const float* __restrict__ bk,
    _Float16* __restrict__ Qt, _Float16* __restrict__ Kt)
{
    const int bt = blockIdx.x;            // 0..9
    const int ch = blockIdx.y;            // 0..63
    const int bb = bt / 5, tt = bt % 5;
    const float* xin = x + ((size_t)bt * 64 + ch) * 4096;

    float qw[9], kw[9];
#pragma unroll
    for (int i = 0; i < 9; ++i) { qw[i] = wq[ch * 9 + i]; kw[i] = wk[ch * 9 + i]; }
    const float qb = bq[ch], kb = bk[ch];

    for (int p = threadIdx.x; p < 4096; p += 256) {
        const int y = p >> 6, xp = p & 63;
        float qa = qb, ka = kb;
#pragma unroll
        for (int ky = 0; ky < 3; ++ky) {
            const int yy = y + ky - 1;
            if (yy < 0 || yy > 63) continue;
#pragma unroll
            for (int kx = 0; kx < 3; ++kx) {
                const int xx = xp + kx - 1;
                if (xx < 0 || xx > 63) continue;
                const float v = xin[yy * 64 + xx];
                qa += v * qw[ky * 3 + kx];
                ka += v * kw[ky * 3 + kx];
            }
        }
        const int tok = tt * 1024 + (y >> 1) * 32 + (xp >> 1);
        const int d   = ch * 4 + (y & 1) * 2 + (xp & 1);
        const size_t o = ((size_t)bb * NTOK + tok) * DH + d;
        Qt[o] = (_Float16)qa;
        Kt[o] = (_Float16)ka;
    }
}

// ---------------------------------------------------------------------------
// Kernel 2: dense 3x3 conv (v), LDS-tiled, fused unfold -> f16 token buffer
// grid: (bt=10, tile=16 of 16x16), block: 256 threads (one pixel each, 64 co)
// ---------------------------------------------------------------------------
__global__ __launch_bounds__(256) void vconv_kernel(
    const float* __restrict__ x,
    const float* __restrict__ wv, const float* __restrict__ bv,
    _Float16* __restrict__ Vt)
{
    __shared__ float patch[18 * 18];
    __shared__ float wsh[64 * 9];

    const int bt = blockIdx.x;
    const int tile = blockIdx.y;
    const int ty0 = (tile >> 2) * 16, tx0 = (tile & 3) * 16;
    const int ly = threadIdx.x >> 4, lx = threadIdx.x & 15;
    const int y = ty0 + ly, xp = tx0 + lx;
    const int bb = bt / 5, tt = bt % 5;
    const float* xin = x + (size_t)bt * 64 * 4096;

    float acc[64];
#pragma unroll
    for (int co = 0; co < 64; ++co) acc[co] = 0.f;

    for (int ci = 0; ci < 64; ++ci) {
        for (int i = threadIdx.x; i < 18 * 18; i += 256) {
            const int py = i / 18 - 1 + ty0, px = i % 18 - 1 + tx0;
            float v = 0.f;
            if (py >= 0 && py < 64 && px >= 0 && px < 64)
                v = xin[(size_t)ci * 4096 + py * 64 + px];
            patch[i] = v;
        }
        for (int i = threadIdx.x; i < 576; i += 256) {
            const int co = i / 9, tap = i % 9;
            wsh[i] = wv[((size_t)co * 64 + ci) * 9 + tap];
        }
        __syncthreads();

        float in[9];
#pragma unroll
        for (int ky = 0; ky < 3; ++ky)
#pragma unroll
            for (int kx = 0; kx < 3; ++kx)
                in[ky * 3 + kx] = patch[(ly + ky) * 18 + (lx + kx)];

#pragma unroll 8
        for (int co = 0; co < 64; ++co) {
            float a = acc[co];
#pragma unroll
            for (int tap = 0; tap < 9; ++tap) a += in[tap] * wsh[co * 9 + tap];
            acc[co] = a;
        }
        __syncthreads();
    }

    const int tok = tt * 1024 + (y >> 1) * 32 + (xp >> 1);
    const int dbase = (y & 1) * 2 + (xp & 1);
    _Float16* vout = Vt + ((size_t)bb * NTOK + tok) * DH;
#pragma unroll 8
    for (int co = 0; co < 64; ++co)
        vout[co * 4 + dbase] = (_Float16)(acc[co] + bv[co]);
}

// ---------------------------------------------------------------------------
// Kernel 3: flash attention with WMMA f16->f32; K tile staged by the
// Tensor Data Mover (tensor_load_to_lds, TENSORcnt).
// grid: (80 row-tiles, 2 batch), block: 128 threads (4 waves, 16 rows each)
// ---------------------------------------------------------------------------
__global__ __launch_bounds__(128) void attn_kernel(
    const _Float16* __restrict__ Qt, const _Float16* __restrict__ Kt,
    const _Float16* __restrict__ Vt, float* __restrict__ Ot)
{
    __shared__ _Float16 ldsK[32 * DH];      // K tile, [j][d] row-major (TDM dest)
    __shared__ _Float16 ldsV[DH * 32];      // V tile, transposed [d][j]
    __shared__ _Float16 ldsP[4][16 * 32];   // per-wave P scratch [m][j]

    const int b    = blockIdx.y;
    const int wave = threadIdx.x >> 5;
    const int lane = threadIdx.x & 31;
    const int m    = lane & 15;             // A-row / B-col within a tile
    const int grp  = lane >> 4;
    const int i0   = blockIdx.x * 64 + wave * 16;

    const _Float16* Q = Qt + (size_t)b * NTOK * DH;
    const _Float16* K = Kt + (size_t)b * NTOK * DH;
    const _Float16* V = Vt + (size_t)b * NTOK * DH;
    float*          O = Ot + (size_t)b * NTOK * DH;

    // Q fragments: 8 chunks of K=32 covering d=256 (held for whole kernel)
    v16h aq[8];
#pragma unroll
    for (int dc = 0; dc < 8; ++dc)
#pragma unroll
        for (int e = 0; e < 16; ++e)
            aq[dc][e] = Q[(size_t)(i0 + m) * DH + dc * 32 + a_frag_k(grp, e)];

    v8f acc[16];                 // 16 rows x 256 d of f32 accumulators
    const v8f zacc = {};
#pragma unroll
    for (int dt = 0; dt < 16; ++dt) acc[dt] = zacc;
    float rowM[8], rowL[8];
#pragma unroll
    for (int r = 0; r < 8; ++r) { rowM[r] = -__builtin_inff(); rowL[r] = 0.f; }

    // TDM descriptor invariants: 1D tile of 32*DH f16 (16 KB), data_size=2B
    const unsigned ldsAddrK = (unsigned)(size_t)(void*)&ldsK[0]; // LDS byte offset

    for (int j0 = 0; j0 < NTOK; j0 += 32) {
        // --- stage K tile [32][256] via Tensor Data Mover (wave 0 only) ---
        if (wave == 0) {
            const unsigned long long ga =
                (unsigned long long)(size_t)(const void*)(K + (size_t)j0 * DH);
            v4u g0;                                    // D# group 0 (ISA 8.3)
            g0[0] = 1u;                                // count=1, user descriptor
            g0[1] = ldsAddrK;                          // lds_addr (bytes)
            g0[2] = (unsigned)ga;                      // global_addr[31:0]
            g0[3] = (unsigned)((ga >> 32) & 0x01FFFFFFu) | (2u << 30); // [56:32]|type=2
            v8u g1;                                    // D# group 1 (ISA 8.4)
            g1[0] = 0x00010000u;                       // data_size=2B, no flags
            g1[1] = 0x2000u << 16;                     // tensor_dim0[15:0]=8192
            g1[2] = 0u | (1u << 16);                   // tensor_dim0 hi=0, tensor_dim1=1
            g1[3] = 0x2000u << 16;                     // tile_dim0=8192
            g1[4] = 1u;                                // tile_dim1=1, tile_dim2=0
            g1[5] = 8192u;                             // tensor_dim0_stride lo
            g1[6] = 0u;                                // stride hi / dim1_stride lo
            g1[7] = 0u;
            asm volatile("tensor_load_to_lds %0, %1\n\t"
                         "s_wait_tensorcnt 0x0"
                         :: "s"(g0), "s"(g1) : "memory");
        }
        // --- stage V tile transposed [256][32] ---
#pragma unroll
        for (int it = 0; it < 8; ++it) {
            const int idx = threadIdx.x + it * 128;   // 0..1023
            const int jj = idx >> 5;                  // token 0..31
            const int dg = idx & 31;                  // group of 8 d's
            const v8h vv = *(const v8h*)(V + (size_t)(j0 + jj) * DH + dg * 8);
#pragma unroll
            for (int q = 0; q < 8; ++q)
                ldsV[(dg * 8 + q) * 32 + jj] = vv[q];
        }
        __syncthreads();

        // --- S = scale * Q K^T : two 16-col subtiles, K-loop over d ---
        v8f s[2];
#pragma unroll
        for (int jt = 0; jt < 2; ++jt) {
            v8f c = {};
#pragma unroll
            for (int dc = 0; dc < 8; ++dc) {
                const v16h bfrag =
                    *(const v16h*)&ldsK[(jt * 16 + m) * DH + dc * 32 + grp * 16];
                c = __builtin_amdgcn_wmma_f32_16x16x32_f16(
                        false, aq[dc], false, bfrag, (short)0, c, false, false);
            }
            s[jt] = c * 0.0625f;   // d^-0.5 = 1/16
        }

        // --- online softmax (rows live on vgpr r, half-wave grp) ---
#pragma unroll
        for (int r = 0; r < 8; ++r) {
            float mx = fmaxf(s[0][r], s[1][r]);
#pragma unroll
            for (int sh = 8; sh >= 1; sh >>= 1)
                mx = fmaxf(mx, __shfl_xor(mx, sh, 32));
            const float nM = fmaxf(rowM[r], mx);
            const float corr = __expf(rowM[r] - nM);
            const float p0 = __expf(s[0][r] - nM);
            const float p1 = __expf(s[1][r] - nM);
            float ps = p0 + p1;
#pragma unroll
            for (int sh = 8; sh >= 1; sh >>= 1)
                ps += __shfl_xor(ps, sh, 32);
            rowL[r] = rowL[r] * corr + ps;
            rowM[r] = nM;
            s[0][r] = p0; s[1][r] = p1;
#pragma unroll
            for (int dt = 0; dt < 16; ++dt) acc[dt][r] *= corr;
        }

        // --- C-layout P -> LDS -> A-layout fragment (f16) ---
        _Float16* pb = &ldsP[wave][0];
#pragma unroll
        for (int jt = 0; jt < 2; ++jt)
#pragma unroll
            for (int r = 0; r < 8; ++r)
                pb[(r + 8 * grp) * 32 + jt * 16 + m] = (_Float16)s[jt][r];
        asm volatile("s_wait_dscnt 0" ::: "memory");  // intra-wave LDS RAW
        v16h ap;
#pragma unroll
        for (int e = 0; e < 16; ++e)
            ap[e] = pb[m * 32 + a_frag_k(grp, e)];

        // --- O += P * V : 16 d-tiles of 16 ---
#pragma unroll
        for (int dt = 0; dt < 16; ++dt) {
            const v16h bfrag = *(const v16h*)&ldsV[(dt * 16 + m) * 32 + grp * 16];
            acc[dt] = __builtin_amdgcn_wmma_f32_16x16x32_f16(
                          false, ap, false, bfrag, (short)0, acc[dt], false, false);
        }
        __syncthreads();
    }

    // --- normalize and store O tokens [b][i][d] f32 ---
#pragma unroll
    for (int r = 0; r < 8; ++r) rowL[r] = 1.0f / rowL[r];
#pragma unroll
    for (int dt = 0; dt < 16; ++dt)
#pragma unroll
        for (int r = 0; r < 8; ++r)
            O[(size_t)(i0 + r + 8 * grp) * DH + dt * 16 + m] = acc[dt][r] * rowL[r];
}

// ---------------------------------------------------------------------------
// Kernel 4: fold (fused gather) + dense 3x3 conv (wc) + bias + residual
// grid: (bt=10, tile=16 of 16x16), block: 256 threads
// ---------------------------------------------------------------------------
__global__ __launch_bounds__(256) void fold_conv_kernel(
    const float* __restrict__ Ot,
    const float* __restrict__ wc, const float* __restrict__ bc,
    const float* __restrict__ x, float* __restrict__ out)
{
    __shared__ float patch[18 * 18];
    __shared__ float wsh[64 * 9];

    const int bt = blockIdx.x;
    const int tile = blockIdx.y;
    const int ty0 = (tile >> 2) * 16, tx0 = (tile & 3) * 16;
    const int ly = threadIdx.x >> 4, lx = threadIdx.x & 15;
    const int y = ty0 + ly, xp = tx0 + lx;
    const int bb = bt / 5, tt = bt % 5;
    const float* Ob = Ot + (size_t)bb * NTOK * DH;

    float acc[64];
#pragma unroll
    for (int co = 0; co < 64; ++co) acc[co] = 0.f;

    for (int ci = 0; ci < 64; ++ci) {
        for (int i = threadIdx.x; i < 18 * 18; i += 256) {
            const int py = i / 18 - 1 + ty0, px = i % 18 - 1 + tx0;
            float v = 0.f;
            if (py >= 0 && py < 64 && px >= 0 && px < 64) {
                const int tok = tt * 1024 + (py >> 1) * 32 + (px >> 1);
                const int d = ci * 4 + (py & 1) * 2 + (px & 1);
                v = Ob[(size_t)tok * DH + d];   // fold = gather from O tokens
            }
            patch[i] = v;
        }
        for (int i = threadIdx.x; i < 576; i += 256) {
            const int co = i / 9, tap = i % 9;
            wsh[i] = wc[((size_t)co * 64 + ci) * 9 + tap];
        }
        __syncthreads();

        float in[9];
#pragma unroll
        for (int ky = 0; ky < 3; ++ky)
#pragma unroll
            for (int kx = 0; kx < 3; ++kx)
                in[ky * 3 + kx] = patch[(ly + ky) * 18 + (lx + kx)];

#pragma unroll 8
        for (int co = 0; co < 64; ++co) {
            float a = acc[co];
#pragma unroll
            for (int tap = 0; tap < 9; ++tap) a += in[tap] * wsh[co * 9 + tap];
            acc[co] = a;
        }
        __syncthreads();
    }

#pragma unroll 8
    for (int co = 0; co < 64; ++co) {
        const size_t o = ((size_t)bt * 64 + co) * 4096 + (size_t)y * 64 + xp;
        out[o] = acc[co] + bc[co] + x[o];
    }
}

// ---------------------------------------------------------------------------
extern "C" void kernel_launch(void* const* d_in, const int* in_sizes, int n_in,
                              void* d_out, int out_size, void* d_ws, size_t ws_size,
                              hipStream_t stream)
{
    const float* x  = (const float*)d_in[0];
    const float* wq = (const float*)d_in[1];
    const float* bq = (const float*)d_in[2];
    const float* wk = (const float*)d_in[3];
    const float* bk = (const float*)d_in[4];
    const float* wv = (const float*)d_in[5];
    const float* bv = (const float*)d_in[6];
    const float* wc = (const float*)d_in[7];
    const float* bc = (const float*)d_in[8];
    float* out = (float*)d_out;

    // workspace: Q,K,V tokens (f16) + O tokens (f32)  ~= 26.2 MB
    const size_t tokElems = (size_t)2 * NTOK * DH;
    _Float16* Qt = (_Float16*)d_ws;
    _Float16* Kt = Qt + tokElems;
    _Float16* Vt = Kt + tokElems;
    float*    Ot = (float*)(Vt + tokElems);

    qk_kernel<<<dim3(10, 64), 256, 0, stream>>>(x, wq, bq, wk, bk, Qt, Kt);
    vconv_kernel<<<dim3(10, 16), 256, 0, stream>>>(x, wv, bv, Vt);
    attn_kernel<<<dim3(80, 2), 128, 0, stream>>>(Qt, Kt, Vt, Ot);
    fold_conv_kernel<<<dim3(10, 16), 256, 0, stream>>>(Ot, wc, bc, x, out);
}
